// ExtraMSAStackBlock_52183852646493
// MI455X (gfx1250) — compile-verified, matrix operand order checked by reading, not run on hardware
//
#include <hip/hip_runtime.h>

typedef __attribute__((ext_vector_type(16))) _Float16 v16h;
typedef __attribute__((ext_vector_type(8)))  float    v8f;

#define LN_EPS 1e-5f

// ============================= elementwise =============================
__global__ void copy_kernel(float* __restrict__ d, const float* __restrict__ s, long n) {
  long i = (long)blockIdx.x * blockDim.x + threadIdx.x;
  if (i < n) d[i] = s[i];
}
__global__ void add_kernel(float* __restrict__ d, const float* __restrict__ s, long n) {
  long i = (long)blockIdx.x * blockDim.x + threadIdx.x;
  if (i < n) d[i] += s[i];
}
__global__ void mul_kernel(float* __restrict__ d, const float* __restrict__ s, long n) {
  long i = (long)blockIdx.x * blockDim.x + threadIdx.x;
  if (i < n) d[i] *= s[i];
}
// zt[i,j,c] = z[j,i,c]
__global__ void transposeT_kernel(float* __restrict__ d, const float* __restrict__ s, int I) {
  long idx = (long)blockIdx.x * blockDim.x + threadIdx.x;
  long n = (long)I * I * 32;
  if (idx >= n) return;
  int c = (int)(idx & 31);
  long t = idx >> 5;
  int j = (int)(t % I);
  int i = (int)(t / I);
  d[idx] = s[(((long)j * I + i) << 5) + c];
}
// z[i,j,c] += out[j,i,c]
__global__ void addT_kernel(float* __restrict__ d, const float* __restrict__ s, int I) {
  long idx = (long)blockIdx.x * blockDim.x + threadIdx.x;
  long n = (long)I * I * 32;
  if (idx >= n) return;
  int c = (int)(idx & 31);
  long t = idx >> 5;
  int j = (int)(t % I);
  int i = (int)(t / I);
  d[idx] += s[(((long)j * I + i) << 5) + c];
}

// LayerNorm over last dim (32). One lane per channel, 8 rows per 256-thread block.
__global__ void ln_kernel(const float* __restrict__ x, const float* __restrict__ g,
                          const float* __restrict__ b, float* __restrict__ y, long rows) {
  long row = (long)blockIdx.x * 8 + (threadIdx.x >> 5);
  int c = threadIdx.x & 31;
  if (row >= rows) return;
  float v = x[row * 32 + c];
  float s = v;
  #pragma unroll
  for (int o = 16; o > 0; o >>= 1) s += __shfl_xor(s, o, 32);
  float mu = s * (1.f / 32.f);
  float d = v - mu;
  float q = d * d;
  #pragma unroll
  for (int o = 16; o > 0; o >>= 1) q += __shfl_xor(q, o, 32);
  float var = q * (1.f / 32.f);
  y[row * 32 + c] = d * rsqrtf(var + LN_EPS) * g[c] + b[c];
}

// qmean[i, hc] = mean over s of q[s,i,hc]
__global__ void qmean_kernel(const float* __restrict__ q, float* __restrict__ qm, int S, int I) {
  int idx = blockIdx.x * blockDim.x + threadIdx.x;
  if (idx >= I * 256) return;
  int i = idx >> 8, hc = idx & 255;
  float s = 0.f;
  for (int t = 0; t < S; ++t) s += q[((long)t * I + i) * 256 + hc];
  qm[idx] = s / (float)S;
}

// Column attention (global over s). grid: (I, H), block: S(=128)
__global__ void colattn_kernel(const float* __restrict__ qmean, const float* __restrict__ k,
                               const float* __restrict__ v, const float* __restrict__ gate,
                               float* __restrict__ o, int S, int I, float scale) {
  __shared__ float red[128];
  int i = blockIdx.x, h = blockIdx.y;
  int t = threadIdx.x;
  float s = 0.f;
  for (int c = 0; c < 32; ++c)
    s += qmean[i * 256 + h * 32 + c] * k[((long)t * I + i) * 32 + c];
  s *= scale;
  red[t] = s; __syncthreads();
  for (int off = 64; off > 0; off >>= 1) { if (t < off) red[t] = fmaxf(red[t], red[t + off]); __syncthreads(); }
  float mx = red[0]; __syncthreads();
  float p = __expf(s - mx);
  red[t] = p; __syncthreads();
  for (int off = 64; off > 0; off >>= 1) { if (t < off) red[t] += red[t + off]; __syncthreads(); }
  float a = p / red[0];
  long base = ((long)t * I + i) * 256 + h * 32;
  long vb = ((long)t * I + i) * 32;
  for (int c = 0; c < 32; ++c) o[base + c] = gate[base + c] * a * v[vb + c];
}

// f32x8 -> low/high half of an f16 fragment (ISA 7.12.2 K-interleave handled by caller)
__device__ __forceinline__ void pack_lo(v16h& f, float4 a, float4 b) {
  f[0] = (_Float16)a.x; f[1] = (_Float16)a.y; f[2] = (_Float16)a.z; f[3] = (_Float16)a.w;
  f[4] = (_Float16)b.x; f[5] = (_Float16)b.y; f[6] = (_Float16)b.z; f[7] = (_Float16)b.w;
}
__device__ __forceinline__ void pack_hi(v16h& f, float4 a, float4 b) {
  f[8]  = (_Float16)a.x; f[9]  = (_Float16)a.y; f[10] = (_Float16)a.z; f[11] = (_Float16)a.w;
  f[12] = (_Float16)b.x; f[13] = (_Float16)b.y; f[14] = (_Float16)b.z; f[15] = (_Float16)b.w;
}

// ============================= generic WMMA GEMM =============================
// C[b1,b2][m,n] = act( cmul * sum_k A[b1,b2][m,k]*B[b1,b2][k,n] + bmul*bias[n] )
// Fully strided so transposes / channel-batched einsums reuse it.
// 128-thread blocks = 4 waves = 2x2 macro-tile of 16x16 WMMA tiles.
// Out-of-range rows/cols are CLAMPED on load (they only feed C entries that are
// never stored), so fragment loads are unpredicated. UNIT_AK is a compile-time
// specialization: unit K-stride on A -> the two contiguous 8-element runs per
// lane become four global_load_b128 (no dual-path in the hot loop).
template <bool UNIT_AK>
__global__ void wmma_gemm_kernel(const float* __restrict__ A, const float* __restrict__ B,
                                 const float* __restrict__ bias, float* __restrict__ C,
                                 int M, int N, int K,
                                 long saR, long saK, long sbK, long sbN, long scR, long scC,
                                 int nB2, long sA1, long sB1, long sC1,
                                 long sA2, long sB2, long sC2,
                                 float cmul, float bmul, int act) {
  int lane = threadIdx.x & 31;
  int wv = threadIdx.x >> 5;
  int bz = blockIdx.z;
  int b1 = bz / nB2, b2 = bz % nB2;
  const float* Ab = A + b1 * sA1 + b2 * sA2;
  const float* Bb = B + b1 * sB1 + b2 * sB2;
  float* Cb = C + b1 * sC1 + b2 * sC2;
  int m0 = (blockIdx.y * 2 + (wv >> 1)) * 16;
  int n0 = (blockIdx.x * 2 + (wv & 1)) * 16;
  int r = lane & 15;
  int kg = (lane >> 4) << 3;   // K interleave base (16-bit A 16x32 layout)
  int row = m0 + r, col = n0 + r;
  long arow = (long)(row < M ? row : M - 1) * saR;
  long bcol = (long)(col < N ? col : N - 1) * sbN;
  v8f acc = {};
  for (int k0 = 0; k0 < K; k0 += 32) {
    if (k0 + 32 < K) {  // uniform branch: prefetch next K-slab
      __builtin_prefetch(Ab + arow + (long)(k0 + 32) * (UNIT_AK ? 1 : saK), 0, 1);
      __builtin_prefetch(Bb + (long)(k0 + 32) * sbK + bcol, 0, 1);
    }
    v16h af, bf;
    if (UNIT_AK) {
      const float* ap = Ab + arow + k0 + kg;
      pack_lo(af, *(const float4*)(ap), *(const float4*)(ap + 4));
      pack_hi(af, *(const float4*)(ap + 16), *(const float4*)(ap + 20));
    } else {
      float ta[16];
      #pragma unroll
      for (int e = 0; e < 16; ++e) {
        int kk = k0 + kg + e + ((e >> 3) << 3);
        ta[e] = Ab[arow + (long)kk * saK];
      }
      #pragma unroll
      for (int e = 0; e < 16; ++e) af[e] = (_Float16)ta[e];
    }
    {
      float tb[16];
      #pragma unroll
      for (int e = 0; e < 16; ++e) {
        int kk = k0 + kg + e + ((e >> 3) << 3);
        tb[e] = Bb[(long)kk * sbK + bcol];
      }
      #pragma unroll
      for (int e = 0; e < 16; ++e) bf[e] = (_Float16)tb[e];
    }
    acc = __builtin_amdgcn_wmma_f32_16x16x32_f16(false, af, false, bf, (short)0, acc, false, false);
  }
  #pragma unroll
  for (int vv = 0; vv < 8; ++vv) {
    int rr = m0 + ((lane >> 4) << 3) + vv;
    int cc = n0 + (lane & 15);
    if (rr < M && cc < N) {
      float x = acc[vv] * cmul;
      if (bias) x += bias[cc] * bmul;
      if (act == 1) x = fmaxf(x, 0.f);
      else if (act == 2) x = 1.f / (1.f + __expf(-x));
      Cb[(long)rr * scR + (long)cc * scC] = x;
    }
  }
}

// ============================= fused attention =============================
// Q,K,V,O laid out [Bo, L, 256] (per-head slice h*32..h*32+31). Two waves per
// block, each owning a 16-query tile + a private 12KB LDS score slab. Scores /
// softmax never touch HBM. bias (optional) indexed [q, j, h], H=8.
__global__ void attn_kernel(const float* __restrict__ Q, const float* __restrict__ K,
                            const float* __restrict__ V, const float* __restrict__ bias,
                            float* __restrict__ O, int L, float scale) {
  __shared__ float sc[2][16 * 192];
  const int HC = 256;
  int lane = threadIdx.x & 31;
  int wv = threadIdx.x >> 5;
  float* scw = sc[wv];
  int q0 = (blockIdx.x * 2 + wv) * 16;
  int h = blockIdx.y;
  long ob = (long)blockIdx.z * L * HC + h * 32;
  int r = lane & 15;
  int kg = (lane >> 4) << 3;
  // --- scores: S = Q(16x32) @ K^T(32xL), one WMMA per 16-wide j tile ---
  v16h qf;
  {
    const float* qp = Q + ob + (long)(q0 + r) * HC + kg;
    pack_lo(qf, *(const float4*)(qp), *(const float4*)(qp + 4));
    pack_hi(qf, *(const float4*)(qp + 16), *(const float4*)(qp + 20));
  }
  for (int j0 = 0; j0 < L; j0 += 16) {
    v16h kf;
    const float* kp = K + ob + (long)(j0 + r) * HC + kg;
    pack_lo(kf, *(const float4*)(kp), *(const float4*)(kp + 4));
    pack_hi(kf, *(const float4*)(kp + 16), *(const float4*)(kp + 20));
    v8f acc = {};
    acc = __builtin_amdgcn_wmma_f32_16x16x32_f16(false, qf, false, kf, (short)0, acc, false, false);
    #pragma unroll
    for (int vv = 0; vv < 8; ++vv) {
      int mr = ((lane >> 4) << 3) + vv;
      scw[mr * 192 + j0 + (lane & 15)] = acc[vv];
    }
  }
  __syncthreads();
  // --- softmax over j (lanes 0..15, one query row each) ---
  if (lane < 16) {
    int qg = q0 + lane;
    float mx = -1e30f;
    for (int j = 0; j < L; ++j) {
      float s = scw[lane * 192 + j] * scale;
      if (bias) s += bias[((long)qg * L + j) * 8 + h];
      scw[lane * 192 + j] = s;
      mx = fmaxf(mx, s);
    }
    float sum = 0.f;
    for (int j = 0; j < L; ++j) {
      float p = __expf(scw[lane * 192 + j] - mx);
      scw[lane * 192 + j] = p;
      sum += p;
    }
    float inv = 1.f / sum;
    for (int j = 0; j < L; ++j) scw[lane * 192 + j] *= inv;
  }
  __syncthreads();
  // --- AV: O = P(16xL) @ V(Lx32) ---
  for (int n0 = 0; n0 < 32; n0 += 16) {
    v8f acc = {};
    for (int k0 = 0; k0 < L; k0 += 32) {
      v16h pf, vf;
      const float* pp = scw + r * 192 + k0 + kg;   // two contiguous runs -> ds_load_b128
      pack_lo(pf, *(const float4*)(pp), *(const float4*)(pp + 4));
      pack_hi(pf, *(const float4*)(pp + 16), *(const float4*)(pp + 20));
      float tv[16];
      #pragma unroll
      for (int e = 0; e < 16; ++e) {
        int kk = k0 + kg + e + ((e >> 3) << 3);
        tv[e] = V[ob + (long)kk * HC + n0 + (lane & 15)];
      }
      #pragma unroll
      for (int e = 0; e < 16; ++e) vf[e] = (_Float16)tv[e];
      acc = __builtin_amdgcn_wmma_f32_16x16x32_f16(false, pf, false, vf, (short)0, acc, false, false);
    }
    #pragma unroll
    for (int vv = 0; vv < 8; ++vv) {
      int rr = q0 + ((lane >> 4) << 3) + vv;
      int cc = n0 + (lane & 15);
      O[ob + (long)rr * HC + cc] = acc[vv];
    }
  }
}

// ============================= host-side orchestration =============================
// Param leaf order: jax tree flatten of setup_inputs() (sorted dict keys, tuple
// order, None dropped).
enum {
  IN_M = 0, IN_Z = 1,
  COL_G_W = 2, COL_K_W, COL_LN_G, COL_LN_B, COL_O_W, COL_O_B, COL_Q_W, COL_V_W,
  MT_I_W = 10, MT_I_B, MT_LN_G, MT_LN_B, MT_O_W, MT_O_B,
  OPM_A_W = 16, OPM_A_B, OPM_B_W, OPM_B_B, OPM_LN_G, OPM_LN_B, OPM_O_W, OPM_O_B,
  PT_I_W = 24, PT_I_B, PT_LN_G, PT_LN_B, PT_O_W, PT_O_B,
  ROW_B_W = 30, ROW_G_W, ROW_K_W, ROW_LN_G, ROW_LN_B, ROW_LNB_G, ROW_LNB_B,
  ROW_O_W, ROW_O_B, ROW_Q_W, ROW_V_W,
  TAE_B_W = 41, TAE_G_W, TAE_G_B, TAE_K_W, TAE_LN_G, TAE_LN_B, TAE_O_W, TAE_O_B, TAE_Q_W, TAE_V_W,
  TAS_B_W = 51, TAS_G_W, TAS_G_B, TAS_K_W, TAS_LN_G, TAS_LN_B, TAS_O_W, TAS_O_B, TAS_Q_W, TAS_V_W,
  TMI_G_W = 61, TMI_G_B, TMI_GA_W, TMI_GA_B, TMI_GB_W, TMI_GB_B, TMI_LN_G, TMI_LN_B,
  TMI_LNO_G, TMI_LNO_B, TMI_O_W, TMI_O_B, TMI_PA_W, TMI_PA_B, TMI_PB_W, TMI_PB_B,
  TMO_G_W = 77, TMO_G_B, TMO_GA_W, TMO_GA_B, TMO_GB_W, TMO_GB_B, TMO_LN_G, TMO_LN_B,
  TMO_LNO_G, TMO_LNO_B, TMO_O_W, TMO_O_B, TMO_PA_W, TMO_PA_B, TMO_PB_W, TMO_PB_B
};

static inline void gemm(hipStream_t st, const float* A, const float* B, const float* bias,
                        float* C, int M, int N, int K,
                        long saR, long saK, long sbK, long sbN, long scR, long scC,
                        int B1 = 1, int B2 = 1,
                        long sA1 = 0, long sB1 = 0, long sC1 = 0,
                        long sA2 = 0, long sB2 = 0, long sC2 = 0,
                        float cmul = 1.f, float bmul = 1.f, int act = 0) {
  dim3 g((N + 31) / 32, (M + 31) / 32, B1 * B2);
  if (saK == 1)
    wmma_gemm_kernel<true><<<g, 128, 0, st>>>(A, B, bias, C, M, N, K, saR, saK, sbK, sbN,
                                              scR, scC, B2, sA1, sB1, sC1, sA2, sB2, sC2,
                                              cmul, bmul, act);
  else
    wmma_gemm_kernel<false><<<g, 128, 0, st>>>(A, B, bias, C, M, N, K, saR, saK, sbK, sbN,
                                               scR, scC, B2, sA1, sB1, sC1, sA2, sB2, sC2,
                                               cmul, bmul, act);
}
// token-major linear: C[T,N] = act(A[T,K] @ W[K,N] + b)
static inline void lin(hipStream_t st, const float* A, const float* W, const float* b,
                       float* C, long T, int N, int K, int act = 0) {
  gemm(st, A, W, b, C, (int)T, N, K, K, 1, N, 1, N, 1, 1, 1, 0, 0, 0, 0, 0, 0, 1.f, 1.f, act);
}
static inline void ew_copy(float* d, const float* s, long n, hipStream_t st) {
  copy_kernel<<<(unsigned)((n + 255) / 256), 256, 0, st>>>(d, s, n);
}
static inline void ew_add(float* d, const float* s, long n, hipStream_t st) {
  add_kernel<<<(unsigned)((n + 255) / 256), 256, 0, st>>>(d, s, n);
}
static inline void ew_mul(float* d, const float* s, long n, hipStream_t st) {
  mul_kernel<<<(unsigned)((n + 255) / 256), 256, 0, st>>>(d, s, n);
}
static inline void run_ln(const float* x, const float* g, const float* b, float* y,
                          long rows, hipStream_t st) {
  ln_kernel<<<(unsigned)((rows + 7) / 8), 256, 0, st>>>(x, g, b, y, rows);
}

extern "C" void kernel_launch(void* const* d_in, const int* in_sizes, int n_in,
                              void* d_out, int out_size, void* d_ws, size_t ws_size,
                              hipStream_t stream) {
  (void)in_sizes; (void)n_in; (void)out_size; (void)ws_size;
  const int S = 128, I = 192;
  const long Tm = (long)S * I, Tz = (long)I * I;
  const float scale = 0.17677669529663689f;  // 32^-0.5
  #define P(i) ((const float*)d_in[i])

  // ---- workspace carve (~250 MB peak, floats) ----
  float* w = (float*)d_ws;
  size_t off = 0;
  auto grab = [&](size_t n) { float* p = w + off; off += n; return p; };
  float* b_ln    = grab(Tz * 32);
  float* b_ln2   = grab(Tz * 32);
  float* b_q     = grab(Tz * 256);
  float* b_k     = grab(Tz * 256);
  float* b_v     = grab(Tz * 256);
  float* b_g     = grab(Tz * 256);
  float* b_o     = grab(Tz * 256);
  float* b_a     = grab(Tz * 32);
  float* b_b     = grab(Tz * 32);
  float* b_x     = grab(Tz * 32);
  float* b_t     = grab(Tz * 128);
  float* b_t2    = grab(Tz * 32);
  float* b_bias  = grab((size_t)I * I * 8);
  float* b_qm    = grab((size_t)I * 256);
  float* b_outer = grab((size_t)16 * I * 1024);
  float* b_zt    = grab(Tz * 32);

  float* m_cur = (float*)d_out;           // residual stream m -> d_out[0 : Tm*32)
  float* z_cur = m_cur + Tm * 32;         // residual stream z -> d_out[Tm*32 : ...)
  ew_copy(m_cur, P(IN_M), Tm * 32, stream);
  ew_copy(z_cur, P(IN_Z), Tz * 32, stream);

  // ================= row attention (m, z-bias) =================
  run_ln(m_cur, P(ROW_LN_G), P(ROW_LN_B), b_ln, Tm, stream);
  run_ln(z_cur, P(ROW_LNB_G), P(ROW_LNB_B), b_ln2, Tz, stream);
  lin(stream, b_ln2, P(ROW_B_W), nullptr, b_bias, Tz, 8, 32);
  lin(stream, b_ln, P(ROW_Q_W), nullptr, b_q, Tm, 256, 32);
  lin(stream, b_ln, P(ROW_K_W), nullptr, b_k, Tm, 256, 32);
  lin(stream, b_ln, P(ROW_V_W), nullptr, b_v, Tm, 256, 32);
  lin(stream, b_ln, P(ROW_G_W), nullptr, b_g, Tm, 256, 32, /*sigmoid*/2);
  attn_kernel<<<dim3(I / 32, 8, S), 64, 0, stream>>>(b_q, b_k, b_v, b_bias, b_o, I, scale);
  ew_mul(b_o, b_g, Tm * 256, stream);
  lin(stream, b_o, P(ROW_O_W), P(ROW_O_B), b_t, Tm, 32, 256);
  ew_add(m_cur, b_t, Tm * 32, stream);

  // ================= column attention (m) =================
  run_ln(m_cur, P(COL_LN_G), P(COL_LN_B), b_ln, Tm, stream);
  lin(stream, b_ln, P(COL_Q_W), nullptr, b_q, Tm, 256, 32);
  qmean_kernel<<<(I * 256 + 255) / 256, 256, 0, stream>>>(b_q, b_qm, S, I);
  lin(stream, b_ln, P(COL_K_W), nullptr, b_k, Tm, 32, 32);
  lin(stream, b_ln, P(COL_V_W), nullptr, b_v, Tm, 32, 32);
  lin(stream, b_ln, P(COL_G_W), nullptr, b_g, Tm, 256, 32, 2);
  colattn_kernel<<<dim3(I, 8), S, 0, stream>>>(b_qm, b_k, b_v, b_g, b_o, S, I, scale);
  lin(stream, b_o, P(COL_O_W), P(COL_O_B), b_t, Tm, 32, 256);
  ew_add(m_cur, b_t, Tm * 32, stream);

  // ================= MSA transition =================
  run_ln(m_cur, P(MT_LN_G), P(MT_LN_B), b_ln, Tm, stream);
  lin(stream, b_ln, P(MT_I_W), P(MT_I_B), b_t, Tm, 128, 32, /*relu*/1);
  lin(stream, b_t, P(MT_O_W), P(MT_O_B), b_t2, Tm, 32, 128);
  ew_add(m_cur, b_t2, Tm * 32, stream);

  // ================= outer product mean (m -> z) =================
  run_ln(m_cur, P(OPM_LN_G), P(OPM_LN_B), b_ln, Tm, stream);
  lin(stream, b_ln, P(OPM_A_W), P(OPM_A_B), b_a, Tm, 32, 32);
  lin(stream, b_ln, P(OPM_B_W), P(OPM_B_B), b_b, Tm, 32, 32);
  {
    const float invn = 1.f / (128.f + 1e-3f);
    for (int i0 = 0; i0 < I; i0 += 16) {
      // outer[i1,j][c,e] = sum_s a[s,i0+i1,c] * b[s,j,e]   (batched 32x32x128)
      gemm(stream, b_a + (long)i0 * 32, b_b, nullptr, b_outer,
           32, 32, S,
           /*saR(c)*/1, /*saK(s)*/(long)I * 32, /*sbK(s)*/(long)I * 32, /*sbN(e)*/1,
           /*scR(c)*/32, /*scC(e)*/1,
           /*B1(i1)*/16, /*B2(j)*/I,
           /*sA1*/32, /*sB1*/0, /*sC1*/(long)I * 1024,
           /*sA2*/0, /*sB2*/32, /*sC2*/1024);
      // project: [16*I, 1024] @ Wo[1024,32], then /(128+eps)
      gemm(stream, b_outer, P(OPM_O_W), P(OPM_O_B), b_t,
           16 * I, 32, 1024, 1024, 1, 32, 1, 32, 1,
           1, 1, 0, 0, 0, 0, 0, 0, invn, invn, 0);
      ew_add(z_cur + (long)i0 * I * 32, b_t, (long)16 * I * 32, stream);
    }
  }

  // ================= triangle multiplication (outgoing, then incoming) =================
  for (int pass = 0; pass < 2; ++pass) {
    const int G_W  = pass == 0 ? TMO_G_W  : TMI_G_W,  G_B  = pass == 0 ? TMO_G_B  : TMI_G_B;
    const int GA_W = pass == 0 ? TMO_GA_W : TMI_GA_W, GA_B = pass == 0 ? TMO_GA_B : TMI_GA_B;
    const int GB_W = pass == 0 ? TMO_GB_W : TMI_GB_W, GB_B = pass == 0 ? TMO_GB_B : TMI_GB_B;
    const int LN_G = pass == 0 ? TMO_LN_G : TMI_LN_G, LN_B = pass == 0 ? TMO_LN_B : TMI_LN_B;
    const int LO_G = pass == 0 ? TMO_LNO_G: TMI_LNO_G,LO_B = pass == 0 ? TMO_LNO_B: TMI_LNO_B;
    const int O_W  = pass == 0 ? TMO_O_W  : TMI_O_W,  O_B  = pass == 0 ? TMO_O_B  : TMI_O_B;
    const int PA_W = pass == 0 ? TMO_PA_W : TMI_PA_W, PA_B = pass == 0 ? TMO_PA_B : TMI_PA_B;
    const int PB_W = pass == 0 ? TMO_PB_W : TMI_PB_W, PB_B = pass == 0 ? TMO_PB_B : TMI_PB_B;
    run_ln(z_cur, P(LN_G), P(LN_B), b_ln, Tz, stream);
    lin(stream, b_ln, P(PA_W), P(PA_B), b_a, Tz, 32, 32);
    lin(stream, b_ln, P(GA_W), P(GA_B), b_x, Tz, 32, 32, 2);
    ew_mul(b_a, b_x, Tz * 32, stream);
    lin(stream, b_ln, P(PB_W), P(PB_B), b_b, Tz, 32, 32);
    lin(stream, b_ln, P(GB_W), P(GB_B), b_x, Tz, 32, 32, 2);
    ew_mul(b_b, b_x, Tz * 32, stream);
    lin(stream, b_ln, P(G_W), P(G_B), b_g, Tz, 32, 32, 2);
    if (pass == 0) {
      // outgoing: x[i,j,c] = sum_k a[i,k,c]*b[j,k,c]  (batched over c)
      gemm(stream, b_a, b_b, nullptr, b_x, I, I, I,
           /*saR(i)*/(long)I * 32, /*saK(k)*/32, /*sbK(k)*/32, /*sbN(j)*/(long)I * 32,
           /*scR(i)*/(long)I * 32, /*scC(j)*/32,
           32, 1, 1, 1, 1, 0, 0, 0);
    } else {
      // incoming: x[i,j,c] = sum_k a[k,i,c]*b[k,j,c]
      gemm(stream, b_a, b_b, nullptr, b_x, I, I, I,
           /*saR(i)*/32, /*saK(k)*/(long)I * 32, /*sbK(k)*/(long)I * 32, /*sbN(j)*/32,
           /*scR(i)*/(long)I * 32, /*scC(j)*/32,
           32, 1, 1, 1, 1, 0, 0, 0);
    }
    run_ln(b_x, P(LO_G), P(LO_B), b_ln2, Tz, stream);
    lin(stream, b_ln2, P(O_W), P(O_B), b_t2, Tz, 32, 32);
    ew_mul(b_t2, b_g, Tz * 32, stream);
    ew_add(z_cur, b_t2, Tz * 32, stream);
  }

  // ================= triangle attention (starting node) =================
  run_ln(z_cur, P(TAS_LN_G), P(TAS_LN_B), b_ln, Tz, stream);
  lin(stream, b_ln, P(TAS_Q_W), nullptr, b_q, Tz, 256, 32);
  lin(stream, b_ln, P(TAS_K_W), nullptr, b_k, Tz, 256, 32);
  lin(stream, b_ln, P(TAS_V_W), nullptr, b_v, Tz, 256, 32);
  lin(stream, b_ln, P(TAS_G_W), P(TAS_G_B), b_g, Tz, 256, 32, 2);
  // bias is broadcast along the softmax axis here -> mathematically a no-op
  attn_kernel<<<dim3(I / 32, 8, I), 64, 0, stream>>>(b_q, b_k, b_v, nullptr, b_o, I, scale);
  ew_mul(b_o, b_g, Tz * 256, stream);
  lin(stream, b_o, P(TAS_O_W), P(TAS_O_B), b_t, Tz, 32, 256);
  ew_add(z_cur, b_t, Tz * 32, stream);

  // ================= triangle attention (ending node: on z^T) =================
  transposeT_kernel<<<(unsigned)((Tz * 32 + 255) / 256), 256, 0, stream>>>(b_zt, z_cur, I);
  run_ln(b_zt, P(TAE_LN_G), P(TAE_LN_B), b_ln, Tz, stream);
  lin(stream, b_ln, P(TAE_Q_W), nullptr, b_q, Tz, 256, 32);
  lin(stream, b_ln, P(TAE_K_W), nullptr, b_k, Tz, 256, 32);
  lin(stream, b_ln, P(TAE_V_W), nullptr, b_v, Tz, 256, 32);
  lin(stream, b_ln, P(TAE_G_W), P(TAE_G_B), b_g, Tz, 256, 32, 2);
  attn_kernel<<<dim3(I / 32, 8, I), 64, 0, stream>>>(b_q, b_k, b_v, nullptr, b_o, I, scale);
  ew_mul(b_o, b_g, Tz * 256, stream);
  lin(stream, b_o, P(TAE_O_W), P(TAE_O_B), b_t, Tz, 32, 256);
  addT_kernel<<<(unsigned)((Tz * 32 + 255) / 256), 256, 0, stream>>>(z_cur, b_t, I);

  // ================= pair transition =================
  run_ln(z_cur, P(PT_LN_G), P(PT_LN_B), b_ln, Tz, stream);
  lin(stream, b_ln, P(PT_I_W), P(PT_I_B), b_t, Tz, 128, 32, 1);
  lin(stream, b_t, P(PT_O_W), P(PT_O_B), b_t2, Tz, 32, 128);
  ew_add(z_cur, b_t2, Tz * 32, stream);
  #undef P
}
